// Sequence_15058155340586
// MI455X (gfx1250) — compile-verified, matrix operand order checked by reading, not run on hardware
//
#include <hip/hip_runtime.h>

// 2-layer LSTM (H=32), teacher-forced + autoregressive, one wave per 16 batch rows.
// Per step: 25x v_wmma_f32_16x16x32_f16 (K=32 == H):
//   8  layer-1 hh   (C = bias + rank-1 input term)
//   16 layer-2 ih+hh (chained through C)
//   1  output projection h2 @ W_lin^T (B non-zero only in column 0)
// D->A relayout via 1KB LDS bounce + ds_load_b128, ordered with s_wait_dscnt.
// Teacher inputs are software-pipelined one step ahead to hide L2 latency.
// Activations use gfx1250 hardware v_tanh_f32 when available.

typedef __attribute__((ext_vector_type(16))) _Float16 v16h;
typedef __attribute__((ext_vector_type(8)))  _Float16 v8h;
typedef __attribute__((ext_vector_type(8)))  float    v8f;
typedef __attribute__((ext_vector_type(4)))  float    v4f;

__device__ __forceinline__ float tanh1(float x) {
#if __has_builtin(__builtin_amdgcn_tanhf)
  return __builtin_amdgcn_tanhf(x);                 // v_tanh_f32
#elif __has_builtin(__builtin_amdgcn_tanh_f32)
  return __builtin_amdgcn_tanh_f32(x);
#else
  float e = __expf(2.0f * x);                       // no IEEE division: raw v_rcp_f32
  return (e - 1.0f) * __builtin_amdgcn_rcpf(e + 1.0f);
#endif
}
__device__ __forceinline__ float sigf(float x) {
  return __builtin_fmaf(0.5f, tanh1(0.5f * x), 0.5f);
}

// K index held by element e of an A/B fragment for lane-half `hi`
// (lanes 0-15 carry K = 0..7,16..23 ; lanes 16-31 carry K = 8..15,24..31)
__device__ __forceinline__ int kmap(int e, int hi) {
  return (e < 8 ? e : e + 8) + 8 * hi;
}

__launch_bounds__(32, 1)
__global__ void lstm_ar_kernel(
    const float* __restrict__ input,
    const float* __restrict__ W_ih1, const float* __restrict__ W_hh1,
    const float* __restrict__ b_ih1, const float* __restrict__ b_hh1,
    const float* __restrict__ W_ih2, const float* __restrict__ W_hh2,
    const float* __restrict__ b_ih2, const float* __restrict__ b_hh2,
    const float* __restrict__ W_lin, const float* __restrict__ b_lin,
    float* __restrict__ out, int T, int F)
{
  const int lane = threadIdx.x & 31;
  const int n    = lane & 15;   // tile column / A-row handled by this lane
  const int hi   = lane >> 4;   // lane half
  const int b0   = blockIdx.x * 16;
  const int TOT  = T + F;

  __shared__ __align__(16) _Float16 hbuf[16 * 32]; // h relayout bounce, row-major [16][32]
  __shared__ __align__(16) float    xbuf[16];      // per-row input / AR feedback

  // ---- stage constant weight B-fragments in VGPRs (live for all steps) ----
  // gates z = h @ W^T : B[k][n] = W[16j+n][k], W is [128,32] row-major.
  v16h wb_hh1[8], wb_ih2[8], wb_hh2[8];
#pragma unroll
  for (int j = 0; j < 8; ++j) {
    const int g = 16 * j + n;
#pragma unroll
    for (int e = 0; e < 16; ++e) {
      const int k = kmap(e, hi);
      wb_hh1[j][e] = (_Float16)W_hh1[g * 32 + k];
      wb_ih2[j][e] = (_Float16)W_ih2[g * 32 + k];
      wb_hh2[j][e] = (_Float16)W_hh2[g * 32 + k];
    }
  }
  // output projection B-fragment: only column n==0 is non-zero
  v16h wb_lin;
#pragma unroll
  for (int e = 0; e < 16; ++e)
    wb_lin[e] = (n == 0) ? (_Float16)W_lin[kmap(e, hi)] : (_Float16)0.0f;

  float bias1[8], wih1v[8];
  v8f bias2v[8], blinv;
#pragma unroll
  for (int j = 0; j < 8; ++j) {
    const int g = 16 * j + n;
    bias1[j] = b_ih1[g] + b_hh1[g];
    wih1v[j] = W_ih1[g];                     // W_ih1 is [128,1]
    const float b2 = b_ih2[g] + b_hh2[g];
#pragma unroll
    for (int r = 0; r < 8; ++r) bias2v[j][r] = b2;
  }
  {
    const float bl = b_lin[0];
#pragma unroll
    for (int r = 0; r < 8; ++r) blinv[r] = bl;
  }

  // recurrent state
  v16h a1 = {}, a2 = {};           // h1,h2 as WMMA A-fragments (f16)
  v8f  c1[2] = {}, c2[2] = {};     // cell states, f32, D-layout chunks (cols 0-15 / 16-31)

  // prime x for t = 0
  if (lane < 16 && T > 0)
    xbuf[lane] = input[(size_t)(b0 + lane) * T];
  float xnext = 0.0f;

  for (int t = 0; t < TOT; ++t) {
    __asm volatile("s_wait_dscnt 0" ::: "memory");
    float xr[8];
    {
      v4f x0 = *(const v4f*)&xbuf[8 * hi];
      v4f x1 = *(const v4f*)&xbuf[8 * hi + 4];
#pragma unroll
      for (int r = 0; r < 4; ++r) { xr[r] = x0[r]; xr[4 + r] = x1[r]; }
    }
    // software-pipelined prefetch of the next teacher input (committed at step end)
    if (lane < 16 && (t + 1) < T)
      xnext = input[(size_t)(b0 + lane) * T + (t + 1)];

    // ---- layer 1: gates = x (.) W_ih1  +  h1 @ W_hh1^T  + bias ----
    v8f g1[8];
#pragma unroll
    for (int j = 0; j < 8; ++j) {
      v8f acc;
#pragma unroll
      for (int r = 0; r < 8; ++r) acc[r] = __builtin_fmaf(xr[r], wih1v[j], bias1[j]);
      g1[j] = __builtin_amdgcn_wmma_f32_16x16x32_f16(
          false, a1, false, wb_hh1[j], (short)0, acc, false, false);
    }
    v8f h1d[2];
#pragma unroll
    for (int ch = 0; ch < 2; ++ch) {
#pragma unroll
      for (int r = 0; r < 8; ++r) {
        float iv = sigf(g1[ch][r]);
        float fv = sigf(g1[2 + ch][r]);
        float gv = tanh1(g1[4 + ch][r]);
        float ov = sigf(g1[6 + ch][r]);
        float cn = __builtin_fmaf(fv, c1[ch][r], iv * gv);
        c1[ch][r]  = cn;
        h1d[ch][r] = ov * tanh1(cn);
      }
    }
    // ---- h1: D-layout -> LDS -> A-fragment ----
#pragma unroll
    for (int ch = 0; ch < 2; ++ch)
#pragma unroll
      for (int r = 0; r < 8; ++r)
        hbuf[(r + 8 * hi) * 32 + ch * 16 + n] = (_Float16)h1d[ch][r];
    __asm volatile("s_wait_dscnt 0" ::: "memory");
    {
      union { v16h v; v8h h[2]; } u;
      u.h[0] = *(const v8h*)&hbuf[n * 32 + 8 * hi];        // K 0..7  (+8*hi)
      u.h[1] = *(const v8h*)&hbuf[n * 32 + 16 + 8 * hi];   // K 16..23 (+8*hi)
      a1 = u.v;
    }

    // ---- layer 2: gates = h1 @ W_ih2^T + h2 @ W_hh2^T + bias (chained C) ----
    v8f g2[8];
#pragma unroll
    for (int j = 0; j < 8; ++j) {
      v8f acc = __builtin_amdgcn_wmma_f32_16x16x32_f16(
          false, a1, false, wb_ih2[j], (short)0, bias2v[j], false, false);
      g2[j] = __builtin_amdgcn_wmma_f32_16x16x32_f16(
          false, a2, false, wb_hh2[j], (short)0, acc, false, false);
    }
    v8f h2d[2];
#pragma unroll
    for (int ch = 0; ch < 2; ++ch) {
#pragma unroll
      for (int r = 0; r < 8; ++r) {
        float iv = sigf(g2[ch][r]);
        float fv = sigf(g2[2 + ch][r]);
        float gv = tanh1(g2[4 + ch][r]);
        float ov = sigf(g2[6 + ch][r]);
        float cn = __builtin_fmaf(fv, c2[ch][r], iv * gv);
        c2[ch][r]  = cn;
        h2d[ch][r] = ov * tanh1(cn);
      }
    }
    // ---- h2: D-layout -> LDS -> A-fragment ----
#pragma unroll
    for (int ch = 0; ch < 2; ++ch)
#pragma unroll
      for (int r = 0; r < 8; ++r)
        hbuf[(r + 8 * hi) * 32 + ch * 16 + n] = (_Float16)h2d[ch][r];
    __asm volatile("s_wait_dscnt 0" ::: "memory");
    {
      union { v16h v; v8h h[2]; } u;
      u.h[0] = *(const v8h*)&hbuf[n * 32 + 8 * hi];
      u.h[1] = *(const v8h*)&hbuf[n * 32 + 16 + 8 * hi];
      a2 = u.v;
    }

    // ---- output projection as a WMMA: D[:,0] = h2 @ W_lin^T + b_lin ----
    v8f od = __builtin_amdgcn_wmma_f32_16x16x32_f16(
        false, a2, false, wb_lin, (short)0, blinv, false, false);
    if (n == 0) {             // lanes 0 (rows 0-7) and 16 (rows 8-15) hold column 0
#pragma unroll
      for (int r = 0; r < 8; ++r) {
        const int m = r + 8 * hi;
        out[(size_t)(b0 + m) * TOT + t] = od[r];
        if (t >= T - 1) xbuf[m] = od[r];  // AR feedback for t+1
      }
    }
    // commit the prefetched teacher input for t+1
    if (lane < 16 && (t + 1) < T)
      xbuf[lane] = xnext;
  }
}

extern "C" void kernel_launch(void* const* d_in, const int* in_sizes, int n_in,
                              void* d_out, int out_size, void* d_ws, size_t ws_size,
                              hipStream_t stream) {
  const float* input = (const float*)d_in[0];
  const float* W_ih1 = (const float*)d_in[1];
  const float* W_hh1 = (const float*)d_in[2];
  const float* b_ih1 = (const float*)d_in[3];
  const float* b_hh1 = (const float*)d_in[4];
  const float* W_ih2 = (const float*)d_in[5];
  const float* W_hh2 = (const float*)d_in[6];
  const float* b_ih2 = (const float*)d_in[7];
  const float* b_hh2 = (const float*)d_in[8];
  const float* W_lin = (const float*)d_in[9];
  const float* b_lin = (const float*)d_in[10];
  // d_in[11] is `future` (device scalar, unreadable during graph capture);
  // derive shapes from sizes: B fixed at 1024 by the source module.
  const int B = 1024;
  const int T = in_sizes[0] / B;        // 999
  const int F = out_size / B - T;       // 1000
  float* out = (float*)d_out;

  dim3 grid(B / 16), block(32);
  hipLaunchKernelGGL(lstm_ar_kernel, grid, block, 0, stream,
                     input, W_ih1, W_hh1, b_ih1, b_hh1,
                     W_ih2, W_hh2, b_ih2, b_hh2, W_lin, b_lin, out, T, F);
}
